// PointerBlock_27633819582599
// MI455X (gfx1250) — matrix operand order repaired and checked by reading, not compiled, not run on hardware
//
#include <hip/hip_runtime.h>
#include <math.h>

// ---------------- problem constants ----------------
constexpr int kB   = 2;
constexpr int kN   = 2048;   // tokens
constexpr int kD   = 1024;   // model dim
constexpr int kH   = 16;     // heads
constexpr int kHD  = 64;     // head dim
constexpr int kTOPK = 8;
constexpr float kSCALE = 0.125f; // 1/sqrt(64)

// ---------------- WMMA fragment types ----------------
typedef __attribute__((ext_vector_type(16))) __bf16 v16bf;
typedef __attribute__((ext_vector_type(8)))  float  v8f;

union FragBF {
    v16bf v;
    unsigned int u[8];
};

__device__ __forceinline__ unsigned short f2bf(float f) {
    unsigned int u = __float_as_uint(f);
    unsigned int r = u + 0x7FFFu + ((u >> 16) & 1u); // RNE
    return (unsigned short)(r >> 16);
}

// CDNA5 async global->LDS copy of 16B, per-lane addresses, ASYNCcnt-tracked.
// INST_OFFSET applies to both global and LDS sides (ISA 15.18.3).
__device__ __forceinline__ void async_copy_b128(unsigned lds_addr, const void* gaddr) {
    asm volatile("global_load_async_to_lds_b128 %0, %1, off"
                 :: "v"(lds_addr), "v"((unsigned long long)(uintptr_t)gaddr)
                 : "memory");
}
__device__ __forceinline__ void async_copy_b128_off16(unsigned lds_addr, const void* gaddr) {
    asm volatile("global_load_async_to_lds_b128 %0, %1, off offset:16"
                 :: "v"(lds_addr), "v"((unsigned long long)(uintptr_t)gaddr)
                 : "memory");
}
__device__ __forceinline__ void wait_async0() {
    asm volatile("s_wait_asynccnt 0" ::: "memory");
}

// ---------------- f32 -> bf16 cast ----------------
__global__ void cast_bf16_kernel(const float* __restrict__ in,
                                 unsigned short* __restrict__ out, int n) {
    int i = blockIdx.x * blockDim.x + threadIdx.x;
    if (i < n) out[i] = f2bf(in[i]);
}

// ---------------- tiled WMMA GEMM: C[m,n] = sum_k A[m,k] * W[n,k] ----------------
// A: bf16 [M x K] row-major, W: bf16 [N x K] row-major (C = A * W^T).
// Block = 128 threads (4 waves) -> 64x64 output tile. Double-buffered LDS
// K-panels staged with CDNA5 async global->LDS DMA; compute on panel i
// overlaps the DMA of panel i+1. Each wave computes a 16x64 strip.
__global__ __launch_bounds__(128) void gemm_tiled_kernel(
        const unsigned short* __restrict__ A,
        const unsigned short* __restrict__ W,
        float* __restrict__ Cf,
        unsigned short* __restrict__ Cbf,
        int M, int N, int K) {
    __shared__ unsigned short As[2][64 * 32];   // 2 x 4 KB
    __shared__ unsigned short Bs[2][64 * 32];   // 2 x 4 KB

    const int t    = threadIdx.x;        // 0..127
    const int wave = t >> 5;             // 0..3
    const int l    = t & 31;
    const int lo   = l & 15;
    const int hi   = l >> 4;

    const int n0 = blockIdx.x * 64;
    const int m0 = blockIdx.y * 64;

    // staging: thread t moves 32B of each panel (row srow, elems scol..scol+15)
    const int srow = t >> 1;             // 0..63
    const int scol = (t & 1) * 16;       // 0 or 16
    const unsigned ldsElem = (unsigned)(srow * 32 + scol) * 2;  // byte offset
    const unsigned aLds[2] = { (unsigned)(uintptr_t)(&As[0][0]) + ldsElem,
                               (unsigned)(uintptr_t)(&As[1][0]) + ldsElem };
    const unsigned bLds[2] = { (unsigned)(uintptr_t)(&Bs[0][0]) + ldsElem,
                               (unsigned)(uintptr_t)(&Bs[1][0]) + ldsElem };
    const unsigned short* gA = A + (size_t)(m0 + srow) * K + scol;
    const unsigned short* gB = W + (size_t)(n0 + srow) * K + scol;

    v8f c[4] = {v8f{}, v8f{}, v8f{}, v8f{}};

    // prologue: stage panel 0 into buffer 0
    async_copy_b128      (aLds[0], gA);
    async_copy_b128_off16(aLds[0], gA);
    async_copy_b128      (bLds[0], gB);
    async_copy_b128_off16(bLds[0], gB);
    wait_async0();
    __syncthreads();

    for (int k0 = 0; k0 < K; k0 += 32) {
        const int cur = (k0 >> 5) & 1;
        const int nxt = cur ^ 1;

        // kick off DMA of the next panel (overlaps with compute below)
        if (k0 + 32 < K) {
            async_copy_b128      (aLds[nxt], gA + k0 + 32);
            async_copy_b128_off16(aLds[nxt], gA + k0 + 32);
            async_copy_b128      (bLds[nxt], gB + k0 + 32);
            async_copy_b128_off16(bLds[nxt], gB + k0 + 32);
        }

        // preload A fragment + all 4 B fragments from the current LDS panel
        FragBF a, b[4];
        const int arow = wave * 16 + lo;
#pragma unroll
        for (int j = 0; j < 8; ++j) {
            int ka = hi * 16 + j * 2;
            a.u[j] = *reinterpret_cast<const unsigned int*>(&As[cur][arow * 32 + ka]);
        }
#pragma unroll
        for (int g = 0; g < 4; ++g) {
            const int bcol = g * 16 + lo;
#pragma unroll
            for (int j = 0; j < 8; ++j)
                b[g].u[j] = *reinterpret_cast<const unsigned int*>(
                                &Bs[cur][bcol * 32 + hi * 16 + 2 * j]);
        }
#pragma unroll
        for (int g = 0; g < 4; ++g)
            c[g] = __builtin_amdgcn_wmma_f32_16x16x32_bf16(false, a.v, false, b[g].v,
                                                           (short)0, c[g], false, false);

        wait_async0();     // next panel's DMA complete (issued before compute)
        __syncthreads();   // all waves done reading cur; nxt writes visible
    }

#pragma unroll
    for (int g = 0; g < 4; ++g) {
#pragma unroll
        for (int j = 0; j < 8; ++j) {
            int m = m0 + wave * 16 + j + hi * 8;
            int n = n0 + g * 16 + lo;
            if (Cf) Cf[(size_t)m * N + n] = c[g][j];
            else    Cbf[(size_t)m * N + n] = f2bf(c[g][j]);
        }
    }
}

// ---------------- fused scores: clip(q.k*scale + alibi) head-mean ----------------
// Q, Kb: bf16 [B, N, D] head-major columns (col = h*64+d). S: f32 [B, N, N].
// One wave computes a 32x32 tile (2x2 fragments) for 2x Q/K reuse.
__global__ void scores_kernel(const unsigned short* __restrict__ Q,
                              const unsigned short* __restrict__ Kb,
                              float* __restrict__ S) {
    const int n0 = blockIdx.x * 32;
    const int m0 = blockIdx.y * 32;
    const int bb = blockIdx.z;
    const int l  = threadIdx.x;
    const int lo = l & 15;
    const int hi = l >> 4;

    const unsigned short* Qb = Q  + (size_t)bb * kN * kD;
    const unsigned short* Kk = Kb + (size_t)bb * kN * kD;
    const unsigned short* arow[2] = { Qb + (size_t)(m0 + lo) * kD,
                                      Qb + (size_t)(m0 + 16 + lo) * kD };
    const unsigned short* brow[2] = { Kk + (size_t)(n0 + lo) * kD,
                                      Kk + (size_t)(n0 + 16 + lo) * kD };

    float acc[2][2][8];
#pragma unroll
    for (int r = 0; r < 2; ++r)
#pragma unroll
        for (int s = 0; s < 2; ++s)
#pragma unroll
            for (int j = 0; j < 8; ++j) acc[r][s][j] = 0.f;

    for (int h = 0; h < kH; ++h) {
        v8f c[2][2] = {{v8f{}, v8f{}}, {v8f{}, v8f{}}};
#pragma unroll
        for (int kk = 0; kk < 2; ++kk) {
            const int k0 = h * kHD + kk * 32;
            FragBF a[2], b[2];
#pragma unroll
            for (int r = 0; r < 2; ++r)
#pragma unroll
                for (int j = 0; j < 8; ++j) {
                    int ka = k0 + hi * 16 + j * 2;
                    a[r].u[j] = *reinterpret_cast<const unsigned int*>(arow[r] + ka);
                    b[r].u[j] = *reinterpret_cast<const unsigned int*>(brow[r] + k0 + hi * 16 + 2 * j);
                }
#pragma unroll
            for (int r = 0; r < 2; ++r)
#pragma unroll
                for (int s = 0; s < 2; ++s)
                    c[r][s] = __builtin_amdgcn_wmma_f32_16x16x32_bf16(
                        false, a[r].v, false, b[s].v, (short)0, c[r][s], false, false);
        }
        const float slope = exp2f(-0.5f * (float)(h + 1));
#pragma unroll
        for (int r = 0; r < 2; ++r)
#pragma unroll
            for (int s = 0; s < 2; ++s)
#pragma unroll
                for (int j = 0; j < 8; ++j) {
                    int m = m0 + r * 16 + j + hi * 8;
                    int n = n0 + s * 16 + lo;
                    float sc = c[r][s][j] * kSCALE - slope * fabsf((float)(m - n));
                    sc = fminf(10.f, fmaxf(-10.f, sc));
                    acc[r][s][j] += sc;
                }
    }

#pragma unroll
    for (int r = 0; r < 2; ++r)
#pragma unroll
        for (int s = 0; s < 2; ++s)
#pragma unroll
            for (int j = 0; j < 8; ++j) {
                int m = m0 + r * 16 + j + hi * 8;
                int n = n0 + s * 16 + lo;
                S[((size_t)bb * kN + m) * kN + n] = acc[r][s][j] * (1.0f / (float)kH);
            }
}

// ---------------- per-row top-8 + softmax (one wave per row) ----------------
__global__ void topk_kernel(const float* __restrict__ S,
                            float* __restrict__ Pws, int* __restrict__ Iws,
                            float* __restrict__ outIdxF, float* __restrict__ outP) {
    const int row = blockIdx.x;                 // b*kN + n
    const float* s = S + (size_t)row * kN;
    const int l = threadIdx.x;                  // 0..31

    float lv[kTOPK];
    int   li[kTOPK];
#pragma unroll
    for (int t = 0; t < kTOPK; ++t) { lv[t] = -3.0e38f; li[t] = 0x7FFFFFFF; }

    for (int j = l; j < kN; j += 32) {
        float v = s[j];
        if (v > lv[kTOPK - 1]) {
            int pos = kTOPK - 1;
#pragma unroll
            for (int q = kTOPK - 1; q > 0; --q) {
                if (lv[q - 1] < v) { lv[q] = lv[q - 1]; li[q] = li[q - 1]; pos = q - 1; }
            }
            lv[pos] = v; li[pos] = j;
        }
    }

    float gv[kTOPK];
    int   gi[kTOPK];
    int ptr = 0;
#pragma unroll
    for (int t = 0; t < kTOPK; ++t) {
        float cand = (ptr < kTOPK) ? lv[ptr] : -3.0e38f;
        int   cidx = (ptr < kTOPK) ? li[ptr] : 0x7FFFFFFF;
        float bv = cand; int bi = cidx;
#pragma unroll
        for (int off = 16; off > 0; off >>= 1) {
            float ov = __shfl_xor(bv, off, 32);
            int   oi = __shfl_xor(bi, off, 32);
            if (ov > bv || (ov == bv && oi < bi)) { bv = ov; bi = oi; }
        }
        gv[t] = bv; gi[t] = bi;
        if (cand == bv && cidx == bi) ++ptr;    // winning lane consumes its head
    }

    // softmax over clip(gv, +-5), computed redundantly by all lanes
    float cv[kTOPK], mx = -3.0e38f;
#pragma unroll
    for (int t = 0; t < kTOPK; ++t) {
        cv[t] = fminf(5.f, fmaxf(-5.f, gv[t]));
        mx = fmaxf(mx, cv[t]);
    }
    float sum = 0.f;
#pragma unroll
    for (int t = 0; t < kTOPK; ++t) { cv[t] = __expf(cv[t] - mx); sum += cv[t]; }
    float inv = 1.0f / sum;

    if (l < kTOPK) {
        float p = cv[l] * inv;
        Pws[(size_t)row * kTOPK + l] = p;
        Iws[(size_t)row * kTOPK + l] = gi[l];
        outIdxF[(size_t)row * kTOPK + l] = (float)gi[l];
        outP[(size_t)row * kTOPK + l]    = p;
    }
}

// ---------------- gather + weighted blend: z_pre = sum_k p_k * V[idx_k] ----------------
__global__ void gather_kernel(const float* __restrict__ V,
                              const float* __restrict__ P,
                              const int* __restrict__ I,
                              unsigned short* __restrict__ Zbf) {
    const int row = blockIdx.x;                 // b*kN + n
    const int b   = row / kN;
    const int t   = threadIdx.x;                // 0..255
    float p[kTOPK]; int idx[kTOPK];
#pragma unroll
    for (int kk = 0; kk < kTOPK; ++kk) {
        p[kk]   = P[(size_t)row * kTOPK + kk];
        idx[kk] = I[(size_t)row * kTOPK + kk];
    }
    const float* Vb = V + (size_t)b * kN * kD;
#pragma unroll
    for (int e = 0; e < kD / 256; ++e) {
        int d = e * 256 + t;
        float acc = 0.f;
#pragma unroll
        for (int kk = 0; kk < kTOPK; ++kk)
            acc += p[kk] * Vb[(size_t)idx[kk] * kD + d];
        Zbf[(size_t)row * kD + d] = f2bf(acc);
    }
}

// ---------------- host-side launch ----------------
extern "C" void kernel_launch(void* const* d_in, const int* in_sizes, int n_in,
                              void* d_out, int out_size, void* d_ws, size_t ws_size,
                              hipStream_t stream) {
    (void)in_sizes; (void)n_in; (void)out_size; (void)ws_size;

    const float* h  = (const float*)d_in[0];
    const float* Wq = (const float*)d_in[1];
    const float* Wk = (const float*)d_in[2];
    const float* Wv = (const float*)d_in[3];
    const float* Wo = (const float*)d_in[4];

    const size_t nTok = (size_t)kB * kN;        // 4096
    const size_t nHD  = nTok * kD;              // 4,194,304
    const size_t nW   = (size_t)kD * kD;        // 1,048,576

    char* ws = (char*)d_ws;
    size_t off = 0;
    unsigned short* hbf = (unsigned short*)(ws + off); off += nHD * 2;
    unsigned short* wqb = (unsigned short*)(ws + off); off += nW * 2;
    unsigned short* wkb = (unsigned short*)(ws + off); off += nW * 2;
    unsigned short* wvb = (unsigned short*)(ws + off); off += nW * 2;
    unsigned short* wob = (unsigned short*)(ws + off); off += nW * 2;
    unsigned short* qbf = (unsigned short*)(ws + off); off += nHD * 2;
    unsigned short* kbf = (unsigned short*)(ws + off); off += nHD * 2;
    float*          vf  = (float*)(ws + off);          off += nHD * 4;
    float*          sco = (float*)(ws + off);          off += (size_t)kB * kN * kN * 4;
    float*          pws = (float*)(ws + off);          off += nTok * kTOPK * 4;
    int*            iws = (int*)(ws + off);            off += nTok * kTOPK * 4;
    unsigned short* zgb = (unsigned short*)(ws + off); off += nHD * 2;

    float* out_z    = (float*)d_out;                      // [B,N,D]
    float* out_idxF = out_z + nHD;                        // [B,N,8] (idx as float)
    float* out_p    = out_idxF + nTok * kTOPK;            // [B,N,8]

    // 1) casts to bf16
    cast_bf16_kernel<<<dim3((unsigned)((nHD + 255) / 256)), 256, 0, stream>>>(h, hbf, (int)nHD);
    cast_bf16_kernel<<<dim3((unsigned)((nW + 255) / 256)), 256, 0, stream>>>(Wq, wqb, (int)nW);
    cast_bf16_kernel<<<dim3((unsigned)((nW + 255) / 256)), 256, 0, stream>>>(Wk, wkb, (int)nW);
    cast_bf16_kernel<<<dim3((unsigned)((nW + 255) / 256)), 256, 0, stream>>>(Wv, wvb, (int)nW);
    cast_bf16_kernel<<<dim3((unsigned)((nW + 255) / 256)), 256, 0, stream>>>(Wo, wob, (int)nW);

    // 2) projections: Q,K (bf16 out), V (f32 out)  -- C = h * W^T, 64x64 LDS-tiled
    dim3 gProj(kD / 64, (unsigned)(nTok / 64));
    gemm_tiled_kernel<<<gProj, 128, 0, stream>>>(hbf, wqb, nullptr, qbf,
                                                 (int)nTok, kD, kD);
    gemm_tiled_kernel<<<gProj, 128, 0, stream>>>(hbf, wkb, nullptr, kbf,
                                                 (int)nTok, kD, kD);
    gemm_tiled_kernel<<<gProj, 128, 0, stream>>>(hbf, wvb, vf, nullptr,
                                                 (int)nTok, kD, kD);

    // 3) fused scores (per-head QK^T, scale, alibi, clip, head-mean), 32x32/wave
    dim3 gSc(kN / 32, kN / 32, kB);
    scores_kernel<<<gSc, 32, 0, stream>>>(qbf, kbf, sco);

    // 4) per-row top-8 + softmax
    topk_kernel<<<dim3((unsigned)nTok), 32, 0, stream>>>(sco, pws, iws, out_idxF, out_p);

    // 5) gather/blend into bf16
    gather_kernel<<<dim3((unsigned)nTok), 256, 0, stream>>>(vf, pws, iws, zgb);

    // 6) final projection: z = z_pre * Wo^T (f32 out, straight to d_out)
    gemm_tiled_kernel<<<gProj, 128, 0, stream>>>(zgb, wob, out_z, nullptr,
                                                 (int)nTok, kD, kD);
}
